// ContrastiveLoss_70351564309266
// MI455X (gfx1250) — compile-verified, matrix operand order checked by reading, not run on hardware
//
#include <hip/hip_runtime.h>

typedef __attribute__((ext_vector_type(16))) __bf16 v16bf;
typedef __attribute__((ext_vector_type(8)))  __bf16 v8bf;
typedef __attribute__((ext_vector_type(4)))  __bf16 v4bf;
typedef __attribute__((ext_vector_type(8)))  float  v8f;
typedef __attribute__((ext_vector_type(4)))  float  v4f;

#define N_ROWS   2048
#define DIM      128
#define M_CAND   16
#define NCOLS    (N_ROWS * M_CAND)          /* 32768 */
#define INV_T    (1.0f / 0.3f)
#define EPS_N    1e-12f
#define LOG2E    1.44269504088896340736f
#define LN2      0.69314718055994530942f

/* d_ws layout (bytes) */
#define A_OFF    0
#define B_OFF    ((size_t)N_ROWS * DIM * 2)               /* 524288  */
#define P_OFF    (B_OFF + (size_t)NCOLS * DIM * 2)        /* 8912896 */
#define NPART    2048

/* ---------------- Stage 1: L2-normalize rows, f32 -> bf16 -------------- */
/* One wave32 per row. Row ids 0..2047 -> anchor/A (pre-scaled by 1/T);
   2048..34815 -> z2/B.                                                    */
__global__ __launch_bounds__(256)
void ContrastiveLoss_normalize_kernel(const float* __restrict__ anchor,
                                      const float* __restrict__ pos,
                                      const float* __restrict__ neg,
                                      __bf16* __restrict__ Abf,
                                      __bf16* __restrict__ Bbf) {
    const int wave = threadIdx.x >> 5;
    const int lane = threadIdx.x & 31;
    const int rid  = blockIdx.x * 8 + wave;        /* 0 .. 34815 */

    const float* src;
    __bf16* dst;
    float post = 1.0f;
    if (rid < N_ROWS) {
        src  = anchor + (size_t)rid * DIM;
        dst  = Abf    + (size_t)rid * DIM;
        post = INV_T;                              /* fold 1/temperature into A */
    } else {
        const int r2 = rid - N_ROWS;               /* column index in B */
        const int j  = r2 >> 4;
        const int m  = r2 & 15;
        src = (m == 0) ? (pos + (size_t)j * DIM)
                       : (neg + ((size_t)j * 15 + (m - 1)) * DIM);
        dst = Bbf + (size_t)r2 * DIM;
    }

    const v4f v = *(const v4f*)(src + lane * 4);
    float s = v.x * v.x + v.y * v.y + v.z * v.z + v.w * v.w;
#pragma unroll
    for (int off = 16; off >= 1; off >>= 1)
        s += __shfl_xor(s, off, 32);

    const float scale = post / fmaxf(__builtin_sqrtf(s), EPS_N);

    v4bf o;
    o.x = (__bf16)(v.x * scale);
    o.y = (__bf16)(v.y * scale);
    o.z = (__bf16)(v.z * scale);
    o.w = (__bf16)(v.w * scale);
    *(v4bf*)(dst + lane * 4) = o;
}

/* B-fragment loader: lane holds column N = lane%16; elements 0..7 carry
   K = 32*kc + 8*half + {0..7}, elements 8..15 carry K = 32*kc+16+8*half+{0..7}
   (mirrors the ISA 16-bit A/B fragment layout).                            */
__device__ __forceinline__ void load_bfrags(const __bf16* __restrict__ bptr,
                                            int half, v16bf bf[4]) {
#pragma unroll
    for (int kc = 0; kc < 4; ++kc) {
        v8bf lo = *(const v8bf*)(bptr + kc * 32 + 8 * half);
        v8bf hi = *(const v8bf*)(bptr + kc * 32 + 16 + 8 * half);
        bf[kc] = __builtin_shufflevector(lo, hi,
                     0, 1, 2, 3, 4, 5, 6, 7, 8, 9, 10, 11, 12, 13, 14, 15);
    }
}

/* -------- Stage 2: fused bf16 WMMA GEMM + BCE-with-logits reduce -------- */
/* grid = (128 N-chunks, 16 M-groups), block = 256 (8 waves).
   Wave w owns M-tile (blockIdx.y*8 + w); walks 16 N-tiles of its N-chunk,
   double-buffering B fragments so loads overlap WMMA + epilogue.          */
__global__ __launch_bounds__(256)
void ContrastiveLoss_gemm_loss_kernel(const __bf16* __restrict__ Abf,
                                      const __bf16* __restrict__ Bbf,
                                      float* __restrict__ partials) {
    __shared__ float red[256];

    const int tid  = threadIdx.x;
    const int wave = tid >> 5;
    const int lane = tid & 31;
    const int half = lane >> 4;
    const int l16  = lane & 15;

    const int mtile = blockIdx.y * 8 + wave;       /* 0..127 */
    const int arow  = mtile * 16 + l16;

    /* A fragments resident for the whole kernel (32 VGPRs) */
    v16bf afrag[4];
    const __bf16* aptr = Abf + (size_t)arow * DIM;
#pragma unroll
    for (int kc = 0; kc < 4; ++kc) {
        v8bf lo = *(const v8bf*)(aptr + kc * 32 + 8 * half);
        v8bf hi = *(const v8bf*)(aptr + kc * 32 + 16 + 8 * half);
        afrag[kc] = __builtin_shufflevector(lo, hi,
                        0, 1, 2, 3, 4, 5, 6, 7, 8, 9, 10, 11, 12, 13, 14, 15);
    }

    float acc = 0.0f;
    const int nt0 = blockIdx.x * 16;
    const __bf16* bbase = Bbf + (size_t)(nt0 * 16 + l16) * DIM;

    v16bf bbuf[2][4];
    load_bfrags(bbase, half, bbuf[0]);

#pragma unroll 2                                   /* ping-pong buffers stay in regs */
    for (int nt = 0; nt < 16; ++nt) {
        const int ntile = nt0 + nt;                /* candidate-owner row j */

        /* issue next tile's B loads before consuming current tile */
        if (nt + 1 < 16)
            load_bfrags(bbase + (size_t)(nt + 1) * 16 * DIM, half,
                        bbuf[(nt + 1) & 1]);

        v8f c = {0.0f, 0.0f, 0.0f, 0.0f, 0.0f, 0.0f, 0.0f, 0.0f};
#pragma unroll
        for (int kc = 0; kc < 4; ++kc)
            c = __builtin_amdgcn_wmma_f32_16x16x32_bf16(
                    false, afrag[kc], false, bbuf[nt & 1][kc],
                    (short)0, c, false, false);

        /* Epilogue: element r of lane L is logits[mtile*16 + r + 8*half][ntile*16+l16],
           already scaled by 1/T. label==1 iff anchor row == ntile.
           max(l,0)+log1p(e^-|l|) via native base-2 exp/log (arg in (1,2], no guards). */
#pragma unroll
        for (int r = 0; r < 8; ++r) {
            const int gm = mtile * 16 + r + 8 * half;
            const float l  = c[r];
            const float e  = __builtin_amdgcn_exp2f(__builtin_fabsf(l) * -LOG2E);
            float t = fmaxf(l, 0.0f) + LN2 * __builtin_amdgcn_logf(1.0f + e);
            if (gm == ntile) t -= l;
            acc += t;
        }
    }

    /* deterministic fixed-order block reduction */
    red[tid] = acc;
    __syncthreads();
#pragma unroll
    for (int s = 128; s > 0; s >>= 1) {
        if (tid < s) red[tid] += red[tid + s];
        __syncthreads();
    }
    if (tid == 0)
        partials[blockIdx.y * gridDim.x + blockIdx.x] = red[0];
}

/* ---------------- Stage 3: deterministic final reduction ---------------- */
__global__ __launch_bounds__(256)
void ContrastiveLoss_final_kernel(const float* __restrict__ partials,
                                  float* __restrict__ out) {
    __shared__ float red[256];
    float s = 0.0f;
    for (int i = threadIdx.x; i < NPART; i += 256)  /* fixed order per lane */
        s += partials[i];
    red[threadIdx.x] = s;
    __syncthreads();
#pragma unroll
    for (int st = 128; st > 0; st >>= 1) {
        if (threadIdx.x < st) red[threadIdx.x] += red[threadIdx.x + st];
        __syncthreads();
    }
    if (threadIdx.x == 0)
        out[0] = red[0] * (1.0f / ((float)N_ROWS * (float)N_ROWS * (float)M_CAND));
}

extern "C" void kernel_launch(void* const* d_in, const int* in_sizes, int n_in,
                              void* d_out, int out_size, void* d_ws, size_t ws_size,
                              hipStream_t stream) {
    const float* anchor = (const float*)d_in[0];   /* [2048,128]     */
    const float* pos    = (const float*)d_in[1];   /* [2048,1,128]   */
    const float* neg    = (const float*)d_in[2];   /* [2048,15,128]  */

    char*   ws       = (char*)d_ws;
    __bf16* Abf      = (__bf16*)(ws + A_OFF);
    __bf16* Bbf      = (__bf16*)(ws + B_OFF);
    float*  partials = (float*)(ws + P_OFF);
    float*  out      = (float*)d_out;

    /* 34816 rows / 8 waves per block */
    ContrastiveLoss_normalize_kernel<<<(N_ROWS + NCOLS) / 8, 256, 0, stream>>>(
        anchor, pos, neg, Abf, Bbf);

    dim3 grid(NCOLS / 16 / 16, (N_ROWS / 16) / 8);   /* (128, 16) */
    ContrastiveLoss_gemm_loss_kernel<<<grid, 256, 0, stream>>>(Abf, Bbf, partials);

    ContrastiveLoss_final_kernel<<<1, 256, 0, stream>>>(partials, out);
}
